// DCNv3_pytorch_5119601017253
// MI455X (gfx1250) — matrix operand order repaired and checked by reading, not compiled, not run on hardware
//
#include <hip/hip_runtime.h>
#include <hip/hip_bf16.h>

typedef __attribute__((ext_vector_type(16))) __bf16 v16bf;
typedef __attribute__((ext_vector_type(8)))  __bf16 v8bf;
typedef __attribute__((ext_vector_type(8)))  float  v8f;

#define C_DIM 1024
#define COUT_DIM 512
#define HH 48
#define WWD 48
#define NB 4
#define GG 4
#define PP 9
#define GCH 256
#define MTOT (NB*HH*WWD)      // 9216
#define OMW 108               // 72 offset + 36 mask cols

#define BM 64
#define BN 256
#define BK 32

union Frag16 { v16bf v; v8bf h[2]; };

// ---- CDNA5 async global->LDS copy (ASYNCcnt-tracked), inline asm = toolchain-portable
__device__ __forceinline__ uint32_t lds_addr(const void* p) {
  // flat LDS aperture keeps the AS3 byte offset in addr[31:0]
  return (uint32_t)(uintptr_t)p;
}
__device__ __forceinline__ void async_b128(uint32_t lds, const void* g) {
  asm volatile("global_load_async_to_lds_b128 %0, %1, off" :: "v"(lds), "v"(g) : "memory");
}
__device__ __forceinline__ void wait_async0() {
#if __has_builtin(__builtin_amdgcn_s_wait_asynccnt)
  __builtin_amdgcn_s_wait_asynccnt(0);
#else
  asm volatile("s_wait_asynccnt 0x0" ::: "memory");
#endif
}

// ---------------- fp32 -> bf16 convert ----------------
__global__ void cvt_f32_bf16_kernel(const float* __restrict__ src,
                                    __bf16* __restrict__ dst, int n) {
  int i = blockIdx.x * blockDim.x + threadIdx.x;
  if (i < n) dst[i] = (__bf16)src[i];
}

// ---------------- concat two bias vectors ----------------
__global__ void concat_bias_kernel(const float* __restrict__ b0, int n0,
                                   const float* __restrict__ b1, int n1,
                                   float* __restrict__ dst) {
  int i = blockIdx.x * blockDim.x + threadIdx.x;
  if (i < n0) dst[i] = b0[i];
  else if (i < n0 + n1) dst[i] = b1[i - n0];
}

// ---------------- bf16 WMMA GEMM: OUT[M,N] = A[M,K] * W[N,K]^T + bias ----------------
// 256 threads = 8 waves; block tile 64x256; wave tile 32x64 (8 WMMA / K-step).
// Double-buffered LDS fed by global_load_async_to_lds_b128; loads use clamped
// (always in-bounds) row indices so the hot loop has no divergent branches.
__global__ __launch_bounds__(256)
void gemm_bf16_wmma(const __bf16* __restrict__ A,
                    const __bf16* __restrict__ Wt,
                    const float* __restrict__ bias,
                    float* __restrict__ OUT,
                    int M, int N, int K)
{
  __shared__ __bf16 As[2][BM][40];   // 80B padded rows: conflict-free, 16B aligned
  __shared__ __bf16 Bs[2][BN][40];

  const int t     = threadIdx.x;
  const int lane  = t & 31;
  const int wave  = t >> 5;
  const int waveM = wave >> 2;       // 0..1 -> 32-row slab
  const int waveN = wave & 3;        // 0..3 -> 64-col slab
  const int half  = lane >> 4;       // K-half selector (ISA 16-bit operand layout)
  const int l16   = lane & 15;

  const int bm = blockIdx.x * BM;
  const int bn = blockIdx.y * BN;

  // loader coords, clamped so every global access is in-bounds
  const int ar  = t >> 2;            // 0..63   A tile row
  const int ac  = (t & 3) << 3;      // 0,8,16,24
  const int br  = t;                 // 0..255  B tile row (= output col)
  const int gmc = min(bm + ar, M - 1);
  const int gnc = min(bn + br, N - 1);
  const __bf16* gA = A  + (size_t)gmc * K + ac;
  const __bf16* gB = Wt + (size_t)gnc * K;

  v8f acc[2][4];
  #pragma unroll
  for (int i = 0; i < 2; i++)
    #pragma unroll
    for (int j = 0; j < 4; j++)
      acc[i][j] = (v8f){0.f,0.f,0.f,0.f,0.f,0.f,0.f,0.f};

  // prologue: stage K-tile 0 into buffer 0
  async_b128(lds_addr(&As[0][ar][ac]), gA);
  #pragma unroll
  for (int j = 0; j < 4; j++)
    async_b128(lds_addr(&Bs[0][br][8 * j]), gB + 8 * j);
  wait_async0();
  __syncthreads();

  for (int kb = 0; kb < K; kb += BK) {
    const int cur = (kb >> 5) & 1;

    if (kb + BK < K) {               // async-prefetch next K tile into other buffer
      async_b128(lds_addr(&As[cur ^ 1][ar][ac]), gA + kb + BK);
      #pragma unroll
      for (int j = 0; j < 4; j++)
        async_b128(lds_addr(&Bs[cur ^ 1][br][8 * j]), gB + kb + BK + 8 * j);
    }

    // A 16x32 frags: lane row M=l16, elems 0-7 -> K=half*8+i, 8-15 -> K=16+half*8+i
    Frag16 a0, a1;
    const int r0 = waveM * 32 + l16;
    a0.h[0] = *(const v8bf*)&As[cur][r0][half * 8];
    a0.h[1] = *(const v8bf*)&As[cur][r0][16 + half * 8];
    a1.h[0] = *(const v8bf*)&As[cur][r0 + 16][half * 8];
    a1.h[1] = *(const v8bf*)&As[cur][r0 + 16][16 + half * 8];

    #pragma unroll
    for (int nsub = 0; nsub < 4; nsub++) {
      // B 32x16 frag: lane col N=l16, elems 0-7 -> K=half*8+i, 8-15 -> K=16+half*8+i
      Frag16 b;
      const int c = waveN * 64 + nsub * 16 + l16;
      b.h[0] = *(const v8bf*)&Bs[cur][c][half * 8];
      b.h[1] = *(const v8bf*)&Bs[cur][c][16 + half * 8];
      acc[0][nsub] = __builtin_amdgcn_wmma_f32_16x16x32_bf16(false, a0.v, false, b.v, (short)0, acc[0][nsub], false, false);
      acc[1][nsub] = __builtin_amdgcn_wmma_f32_16x16x32_bf16(false, a1.v, false, b.v, (short)0, acc[1][nsub], false, false);
    }

    wait_async0();                   // own async writes to next buffer landed
    __syncthreads();                 // everyone's landed; everyone done reading cur
  }

  // C/D layout: VGPR r -> M = r + half*8 ; N = l16
  #pragma unroll
  for (int msub = 0; msub < 2; msub++) {
    #pragma unroll
    for (int nsub = 0; nsub < 4; nsub++) {
      const int n = bn + waveN * 64 + nsub * 16 + l16;
      const float bvv = (n < N) ? bias[n] : 0.0f;
      const int mb = bm + waveM * 32 + msub * 16 + half * 8;
      #pragma unroll
      for (int r = 0; r < 8; r++) {
        const int m = mb + r;
        if (m < M && n < N) OUT[(size_t)m * N + n] = acc[msub][nsub][r] + bvv;
      }
    }
  }
}

// ---------------- fused depthwise 3x3 conv + LayerNorm + exact GELU -> bf16 ----------------
__global__ __launch_bounds__(256)
void dwconv_ln_gelu(const float* __restrict__ input, const float* __restrict__ wdw,
                    const float* __restrict__ bdw, const float* __restrict__ gamma,
                    const float* __restrict__ beta, __bf16* __restrict__ x1b)
{
  __shared__ float red1[256];
  __shared__ float red2[256];
  __shared__ float stats[2];
  const int m  = blockIdx.x;
  const int n  = m / (HH * WWD);
  const int hw = m % (HH * WWD);
  const int h  = hw / WWD;
  const int w  = hw % WWD;
  const int t  = threadIdx.x;

  float v[4];
  float s1 = 0.f, s2 = 0.f;
  #pragma unroll
  for (int i = 0; i < 4; i++) {
    const int c = t + i * 256;
    float a = bdw[c];
    #pragma unroll
    for (int kh = 0; kh < 3; kh++) {
      const int h2 = h + kh - 1;
      if (h2 < 0 || h2 >= HH) continue;
      #pragma unroll
      for (int kw = 0; kw < 3; kw++) {
        const int w2 = w + kw - 1;
        if (w2 < 0 || w2 >= WWD) continue;
        a += input[(((size_t)n * HH + h2) * WWD + w2) * C_DIM + c] * wdw[c * 9 + kh * 3 + kw];
      }
    }
    v[i] = a; s1 += a; s2 += a * a;
  }
  red1[t] = s1; red2[t] = s2;
  __syncthreads();
  for (int s = 128; s > 0; s >>= 1) {
    if (t < s) { red1[t] += red1[t + s]; red2[t] += red2[t + s]; }
    __syncthreads();
  }
  if (t == 0) {
    const float mu  = red1[0] * (1.0f / 1024.0f);
    const float var = red2[0] * (1.0f / 1024.0f) - mu * mu;
    stats[0] = mu; stats[1] = rsqrtf(var + 1e-6f);
  }
  __syncthreads();
  const float mu = stats[0], rs = stats[1];
  #pragma unroll
  for (int i = 0; i < 4; i++) {
    const int c = t + i * 256;
    const float y = (v[i] - mu) * rs * gamma[c] + beta[c];
    const float g = 0.5f * y * (1.0f + erff(y * 0.70710678118654752f));
    x1b[(size_t)m * C_DIM + c] = (__bf16)g;
  }
}

// ---------------- softmax over 9 mask logits per (pixel, group), in place ----------------
__global__ void softmax_mask(float* __restrict__ om) {
  const int idx = blockIdx.x * blockDim.x + threadIdx.x;  // m*4 + g
  if (idx >= MTOT * GG) return;
  const int m = idx >> 2, g = idx & 3;
  float* p = om + (size_t)m * OMW + 72 + g * 9;
  float mx = p[0];
  #pragma unroll
  for (int i = 1; i < 9; i++) mx = fmaxf(mx, p[i]);
  float e[9]; float s = 0.f;
  #pragma unroll
  for (int i = 0; i < 9; i++) { e[i] = expf(p[i] - mx); s += e[i]; }
  const float inv = 1.0f / s;
  #pragma unroll
  for (int i = 0; i < 9; i++) p[i] = e[i] * inv;
}

// ---------------- DCNv3 core: bilinear gather + mask-weighted sum -> bf16 ----------------
__global__ __launch_bounds__(256)
void dcn_core(const float* __restrict__ xproj, const float* __restrict__ om,
              __bf16* __restrict__ dcnb)
{
  __shared__ float sOM[OMW];
  const int m  = blockIdx.x;
  const int n  = m / (HH * WWD);
  const int hw = m % (HH * WWD);
  const int h  = hw / WWD;
  const int w  = hw % WWD;
  const int t  = threadIdx.x;
  if (t < OMW) sOM[t] = om[(size_t)m * OMW + t];
  __syncthreads();

  const float* base = xproj + (size_t)n * HH * WWD * C_DIM;
  #pragma unroll
  for (int g = 0; g < GG; g++) {
    float acc = 0.f;
    const int coff = g * GCH + t;
    #pragma unroll
    for (int p = 0; p < PP; p++) {
      const int gx = (p / 3) - 1;          // kw-major flatten, x varies slowest
      const int gy = (p % 3) - 1;
      const float ox  = sOM[g * 18 + p * 2 + 0];
      const float oy  = sOM[g * 18 + p * 2 + 1];
      const float msk = sOM[72 + g * 9 + p];
      // padded-frame coord minus 1 == unpadded coord; zero-pad taps contribute 0
      const float qx = (float)(w + gx) + ox;
      const float qy = (float)(h + gy) + oy;
      const float fx0 = floorf(qx), fy0 = floorf(qy);
      const float wx1 = qx - fx0, wy1 = qy - fy0;
      const float wx0 = 1.f - wx1, wy0 = 1.f - wy1;
      const int x0 = (int)fx0, y0 = (int)fy0;
      auto tap = [&](int xi, int yi, float wgt) -> float {
        if (xi < 0 || xi >= WWD || yi < 0 || yi >= HH) return 0.f;
        return wgt * base[((size_t)yi * WWD + xi) * C_DIM + coff];
      };
      const float s = tap(x0, y0,     wx0 * wy0) + tap(x0 + 1, y0,     wx1 * wy0)
                    + tap(x0, y0 + 1, wx0 * wy1) + tap(x0 + 1, y0 + 1, wx1 * wy1);
      acc += msk * s;
    }
    dcnb[(size_t)m * C_DIM + coff] = (__bf16)acc;
  }
}

// ---------------- host launcher ----------------
extern "C" void kernel_launch(void* const* d_in, const int* in_sizes, int n_in,
                              void* d_out, int out_size, void* d_ws, size_t ws_size,
                              hipStream_t stream) {
  const float* input = (const float*)d_in[0];
  const float* w_in  = (const float*)d_in[1];
  const float* b_in  = (const float*)d_in[2];
  const float* w_dw  = (const float*)d_in[3];
  const float* b_dw  = (const float*)d_in[4];
  const float* ln_g  = (const float*)d_in[5];
  const float* ln_b  = (const float*)d_in[6];
  const float* w_off = (const float*)d_in[7];
  const float* b_off = (const float*)d_in[8];
  const float* w_msk = (const float*)d_in[9];
  const float* b_msk = (const float*)d_in[10];
  const float* w_out = (const float*)d_in[11];
  const float* b_out = (const float*)d_in[12];
  float* out = (float*)d_out;

  char* ws = (char*)d_ws;
  size_t off = 0;
  auto alloc = [&](size_t bytes) -> void* {
    void* p = ws + off;
    off = (off + bytes + 255) & ~(size_t)255;
    return p;
  };

  __bf16* inA    = (__bf16*)alloc((size_t)MTOT * C_DIM * 2);       // input bf16
  __bf16* wInB   = (__bf16*)alloc((size_t)C_DIM * C_DIM * 2);      // w_in_proj bf16
  __bf16* wOMB   = (__bf16*)alloc((size_t)OMW * C_DIM * 2);        // [w_offset; w_mask] bf16
  __bf16* wOutB  = (__bf16*)alloc((size_t)COUT_DIM * C_DIM * 2);   // w_out bf16
  float*  omBias = (float*)alloc(OMW * 4);
  float*  xproj  = (float*)alloc((size_t)MTOT * C_DIM * 4);        // in_proj output (fp32, sampled)
  __bf16* x1b    = (__bf16*)alloc((size_t)MTOT * C_DIM * 2);       // dw+LN+GELU output
  float*  om     = (float*)alloc((size_t)MTOT * OMW * 4);          // offsets (0..71) + mask (72..107)
  __bf16* dcnb   = (__bf16*)alloc((size_t)MTOT * C_DIM * 2);       // DCN output bf16
  (void)ws_size; (void)n_in; (void)in_sizes; (void)out_size;

  // 1) convert inputs / weights to bf16
  {
    int n = MTOT * C_DIM;
    cvt_f32_bf16_kernel<<<(n + 255) / 256, 256, 0, stream>>>(input, inA, n);
    n = C_DIM * C_DIM;
    cvt_f32_bf16_kernel<<<(n + 255) / 256, 256, 0, stream>>>(w_in, wInB, n);
    n = 72 * C_DIM;
    cvt_f32_bf16_kernel<<<(n + 255) / 256, 256, 0, stream>>>(w_off, wOMB, n);
    n = 36 * C_DIM;
    cvt_f32_bf16_kernel<<<(n + 255) / 256, 256, 0, stream>>>(w_msk, wOMB + (size_t)72 * C_DIM, n);
    n = COUT_DIM * C_DIM;
    cvt_f32_bf16_kernel<<<(n + 255) / 256, 256, 0, stream>>>(w_out, wOutB, n);
    concat_bias_kernel<<<1, 128, 0, stream>>>(b_off, 72, b_msk, 36, omBias);
  }

  // 2) in_proj GEMM: xproj = input @ w_in^T + b_in  (9216x1024x1024)
  {
    dim3 grid(MTOT / BM, (C_DIM + BN - 1) / BN);
    gemm_bf16_wmma<<<grid, 256, 0, stream>>>(inA, wInB, b_in, xproj, MTOT, C_DIM, C_DIM);
  }

  // 3) depthwise conv + LN + GELU -> x1 (bf16)
  dwconv_ln_gelu<<<MTOT, 256, 0, stream>>>(input, w_dw, b_dw, ln_g, ln_b, x1b);

  // 4) offset/mask GEMM: om = x1 @ [w_off;w_msk]^T + [b_off;b_msk]  (9216x108x1024)
  {
    dim3 grid(MTOT / BM, (OMW + BN - 1) / BN);
    gemm_bf16_wmma<<<grid, 256, 0, stream>>>(x1b, wOMB, omBias, om, MTOT, OMW, C_DIM);
  }

  // 5) softmax over 9 points per (pixel, group), in place on mask cols
  softmax_mask<<<(MTOT * GG + 255) / 256, 256, 0, stream>>>(om);

  // 6) DCN bilinear gather core -> dcnb (bf16)
  dcn_core<<<MTOT, 256, 0, stream>>>(xproj, om, dcnb);

  // 7) out GEMM: out = dcnb @ w_out^T + b_out  (9216x512x1024)
  {
    dim3 grid(MTOT / BM, (COUT_DIM + BN - 1) / BN);
    gemm_bf16_wmma<<<grid, 256, 0, stream>>>(dcnb, wOutB, b_out, out, MTOT, COUT_DIM, C_DIM);
  }
}
